// TextEncoder_78391743086941
// MI455X (gfx1250) — matrix operand order repaired and checked by reading, not compiled, not run on hardware
//
#include <hip/hip_runtime.h>
#include <hip/hip_bf16.h>

// ---------------------------------------------------------------------------
// Model constants
// ---------------------------------------------------------------------------
#define DMODEL 1024
#define FFDIM  4096
#define NLAYER 6
#define NHEAD  8
#define BATCH  256
#define SEQ    60
#define DKHEAD 128               // DMODEL / NHEAD
#define MROWS  (BATCH * SEQ)     // 15360

typedef __bf16 bf16_t;
typedef __attribute__((ext_vector_type(16))) __bf16 v16bf;
typedef __attribute__((ext_vector_type(8)))  __bf16 v8bf;
typedef __attribute__((ext_vector_type(8)))  float  v8f;
typedef __attribute__((ext_vector_type(4)))  unsigned tdm_u32x4;
typedef __attribute__((ext_vector_type(8)))  int      tdm_i32x8;
typedef __attribute__((ext_vector_type(4)))  int      tdm_i32x4;

// ---------------------------------------------------------------------------
// Tensor Data Mover: issue a 2D tile load (128 rows x 32 bf16 cols) into LDS.
// Descriptor packing per CDNA5 ISA 8.3 (group0) / 8.4 (group1):
//   group0: count=1 | lds_addr | global_addr[31:0] | global_addr[56:32],type=2
//   group1: data_size=1(2B); tensor_dim0=K, tensor_dim1=M;
//           tile_dim0=32, tile_dim1=128; tensor_dim0_stride=K
// This toolchain's builtin is the 6-arg form:
//   (u32x4 g0, i32x8 g1, i32x4 g2, i32x4 g3, i32x8 gx, i32 cpol)
// ---------------------------------------------------------------------------
__device__ inline void tdm_load_a_tile(const bf16_t* gptr, unsigned ldsAddr,
                                       int K, int M) {
    unsigned long long ga = (unsigned long long)(uintptr_t)gptr;
    tdm_u32x4 g0;
    g0[0] = 1u;                                          // count = 1
    g0[1] = ldsAddr;                                     // lds_addr (bytes)
    g0[2] = (unsigned)ga;                                // global_addr lo
    g0[3] = (unsigned)((ga >> 32) & 0x1FFFFFFu) | (2u << 30); // hi | type=2
    tdm_i32x8 g1;
    g1[0] = (int)(1u << 16);                             // data_size = 2 bytes
    g1[1] = (int)(((unsigned)K & 0xFFFFu) << 16);        // tensor_dim0 lo16
    g1[2] = (int)(((unsigned)K >> 16) |
                  (((unsigned)M & 0xFFFFu) << 16));      // dim0 hi | dim1 lo
    g1[3] = (int)(((unsigned)M >> 16) | (32u << 16));    // dim1 hi | tile_dim0
    g1[4] = (int)128u;                                   // tile_dim1 (=BM)
    g1[5] = (int)(unsigned)K;                            // dim0_stride lo32
    g1[6] = 0;                                           // stride hi | d1s lo
    g1[7] = 0;
    tdm_i32x4 gz4 = {0, 0, 0, 0};                        // 2D: groups 2/3 off
    tdm_i32x8 gz8 = {0, 0, 0, 0, 0, 0, 0, 0};
    __builtin_amdgcn_tensor_load_to_lds(g0, g1, gz4, gz4, gz8, 0);
}

// ---------------------------------------------------------------------------
// fp32 -> bf16 conversion (plain)
// ---------------------------------------------------------------------------
__global__ void cvt_f32_bf16_kernel(const float* __restrict__ src,
                                    bf16_t* __restrict__ dst, size_t n) {
    size_t i = (size_t)blockIdx.x * blockDim.x + threadIdx.x;
    if (i < n) dst[i] = (bf16_t)src[i];
}

// ---------------------------------------------------------------------------
// Per-layer weight transpose + convert: W[l][K][N] f32 -> WT[l][N][K] bf16.
// LDS-tiled 32x32 so both global streams are coalesced. grid.z = layer.
// ---------------------------------------------------------------------------
__global__ void transpose_cvt_kernel(const float* __restrict__ W,
                                     bf16_t* __restrict__ WT, int K, int N) {
    __shared__ float tile[32][33];
    const size_t zoff = (size_t)blockIdx.z * K * N;
    const int kb = blockIdx.x * 32;
    const int nb = blockIdx.y * 32;
    const int tx = threadIdx.x;              // 0..31
    const int ty = threadIdx.y;              // 0..7
#pragma unroll
    for (int r = 0; r < 32; r += 8)
        tile[ty + r][tx] = W[zoff + (size_t)(kb + ty + r) * N + nb + tx];
    __syncthreads();
#pragma unroll
    for (int r = 0; r < 32; r += 8)
        WT[zoff + (size_t)(nb + ty + r) * K + kb + tx] =
            (bf16_t)tile[tx][ty + r];
}

// ---------------------------------------------------------------------------
// Embedding * sqrt(D) + sinusoidal positional encoding -> X (fp32)
// ---------------------------------------------------------------------------
__global__ void embed_kernel(const int* __restrict__ src,
                             const float* __restrict__ embed,
                             float* __restrict__ X) {
    size_t i = (size_t)blockIdx.x * blockDim.x + threadIdx.x;
    if (i >= (size_t)MROWS * DMODEL) return;
    int d      = (int)(i % DMODEL);
    size_t row = i / DMODEL;
    int s      = (int)(row % SEQ);
    int tok    = src[row];
    float val  = embed[(size_t)tok * DMODEL + d] * 32.0f; // sqrt(1024)
    int i2     = d & ~1;
    float div  = __expf(-9.210340371976184f * (float)i2 * (1.0f / DMODEL));
    float ang  = (float)s * div;
    val       += (d & 1) ? __cosf(ang) : __sinf(ang);
    X[i] = val;
}

// ---------------------------------------------------------------------------
// LayerNorm (ddof=1, matches reference). Writes bf16 (layers) or fp32 (final).
// ---------------------------------------------------------------------------
template <bool FINAL_F32>
__global__ void ln_kernel(const float* __restrict__ x,
                          const float* __restrict__ g,
                          const float* __restrict__ b,
                          bf16_t* __restrict__ ybf,
                          float* __restrict__ yf) {
    __shared__ float red[256];
    __shared__ float s_mu, s_inv;
    const int row = blockIdx.x;
    const int t   = threadIdx.x;
    const float* xr = x + (size_t)row * DMODEL;

    float s = 0.f;
#pragma unroll
    for (int i = 0; i < 4; ++i) s += xr[t + 256 * i];
    red[t] = s;
    __syncthreads();
    for (int off = 128; off > 0; off >>= 1) {
        if (t < off) red[t] += red[t + off];
        __syncthreads();
    }
    if (t == 0) s_mu = red[0] * (1.0f / DMODEL);
    __syncthreads();
    const float mu = s_mu;

    float s2 = 0.f;
#pragma unroll
    for (int i = 0; i < 4; ++i) {
        float d = xr[t + 256 * i] - mu;
        s2 += d * d;
    }
    __syncthreads();
    red[t] = s2;
    __syncthreads();
    for (int off = 128; off > 0; off >>= 1) {
        if (t < off) red[t] += red[t + off];
        __syncthreads();
    }
    if (t == 0) {
        float sd = sqrtf(red[0] * (1.0f / (DMODEL - 1)));
        s_inv = 1.0f / (sd + 1e-6f);
    }
    __syncthreads();
    const float inv = s_inv;

#pragma unroll
    for (int i = 0; i < 4; ++i) {
        int c = t + 256 * i;
        float yv = g[c] * (xr[c] - mu) * inv + b[c];
        if (FINAL_F32) yf[(size_t)row * DMODEL + c] = yv;
        else           ybf[(size_t)row * DMODEL + c] = (bf16_t)yv;
    }
}

// ---------------------------------------------------------------------------
// WMMA bf16 GEMM:  C(MxN) = A(MxK,row) * W(KxN) + bias,  W given transposed
// as WT(NxK,row).  Block = 256 threads = 8 waves, macro-tile 128(M) x 64(N);
// wave w computes rows [w*16, w*16+16) x 64 cols = 4 WMMA tiles sharing one
// A fragment.  A tile (128x32) is DMA'd into double-buffered LDS by the
// Tensor Data Mover (wave 0 issues, s_wait_tensorcnt + barrier hand-off);
// B fragments are contiguous 16B per-lane loads straight from WT (L2).
// FLAGS: 1 = ReLU, 2 = residual fp32 (resOut = resIn + C), 4 = write bf16
// ---------------------------------------------------------------------------
template <int FLAGS>
__global__ void gemm_wmma_bf16_kernel(const bf16_t* __restrict__ A,
                                      const bf16_t* __restrict__ WT,
                                      const float* __restrict__ bias,
                                      const float* __restrict__ resIn,
                                      float* __restrict__ resOut,
                                      bf16_t* __restrict__ outBf,
                                      int M, int N, int K) {
    __shared__ __align__(16) bf16_t lA[2][128][32];      // 2 x 8 KB

    const int t       = threadIdx.x;
    const int lane    = t & 31;
    const int w       = t >> 5;                 // wave id 0..7
    const int rowBase = blockIdx.y * 128;
    const int colBase = blockIdx.x * 64;
    const int nslab   = K >> 5;                 // K / 32

    v8f acc[4] = {};

    // prologue: DMA slab 0 into buffer 0
    if (w == 0) {
        tdm_load_a_tile(A + (size_t)rowBase * K,
                        (unsigned)(uintptr_t)&lA[0][0][0], K, M);
    }

    for (int s = 0; s < nslab; ++s) {
        const int cur = s & 1;
        const int k0  = s << 5;
        if (w == 0) {
            if (s + 1 < nslab) {
                tdm_load_a_tile(A + (size_t)rowBase * K + (k0 + 32),
                                (unsigned)(uintptr_t)&lA[cur ^ 1][0][0], K, M);
                __builtin_amdgcn_s_wait_tensorcnt(1);  // slab s landed
            } else {
                __builtin_amdgcn_s_wait_tensorcnt(0);
            }
        }
        __syncthreads();   // LDS buffer `cur` visible to all waves

        // --- A fragment (wave32 16x32 bf16 layout)
        const int aRow = w * 16 + (lane & 15);
        const int ksel = (lane >> 4) * 8;       // 0 or 8
        v8bf alo = *(const v8bf*)&lA[cur][aRow][ksel];
        v8bf ahi = *(const v8bf*)&lA[cur][aRow][ksel + 16];
        v16bf a;
#pragma unroll
        for (int i = 0; i < 8; ++i) { a[i] = alo[i]; a[8 + i] = ahi[i]; }

        // --- 4 B fragments straight from WT (lane holds one N-column,
        //     K values kb..kb+7 and kb+16..kb+23) + 4 WMMAs reusing `a`
        const int kb = (lane >> 4) * 8;         // 0 or 8
#pragma unroll
        for (int n = 0; n < 4; ++n) {
            const int colN = colBase + n * 16 + (lane & 15);
            const bf16_t* wp = WT + (size_t)colN * K + k0 + kb;
            v8bf blo = *(const v8bf*)wp;
            v8bf bhi = *(const v8bf*)(wp + 16);
            v16bf bm;
#pragma unroll
            for (int i = 0; i < 8; ++i) { bm[i] = blo[i]; bm[8 + i] = bhi[i]; }
            acc[n] = __builtin_amdgcn_wmma_f32_16x16x32_bf16(
                         false, a, false, bm, (short)0, acc[n], false, false);
        }
        __syncthreads();   // done reading `cur` before it is DMA-overwritten
    }

    // --- epilogue: C layout -> VGPR r: lanes0-15 M=r, lanes16-31 M=r+8
    const int half = lane >> 4;
#pragma unroll
    for (int n = 0; n < 4; ++n) {
        const int colN = colBase + n * 16 + (lane & 15);
        const float bv = bias[colN];
#pragma unroll
        for (int r = 0; r < 8; ++r) {
            int rowM = rowBase + w * 16 + r + half * 8;
            float val = acc[n][r] + bv;
            if (FLAGS & 1) val = fmaxf(val, 0.0f);
            if (FLAGS & 2) {
                size_t idx = (size_t)rowM * N + colN;
                resOut[idx] = resIn[idx] + val;
            }
            if (FLAGS & 4) {
                outBf[(size_t)rowM * N + colN] = (bf16_t)val;
            }
        }
    }
}

// ---------------------------------------------------------------------------
// Attention: one block per (b,h); 64 threads; K/V/Q tiles staged in LDS.
// ---------------------------------------------------------------------------
__global__ void attn_kernel(const bf16_t* __restrict__ q,
                            const bf16_t* __restrict__ k,
                            const bf16_t* __restrict__ v,
                            const int* __restrict__ src,
                            bf16_t* __restrict__ o) {
    __shared__ __align__(16) bf16_t lq[SEQ][DKHEAD];
    __shared__ __align__(16) bf16_t lk[SEQ][DKHEAD];
    __shared__ __align__(16) bf16_t lv[SEQ][DKHEAD];
    __shared__ float mk[SEQ];

    const int b = blockIdx.x / NHEAD;
    const int h = blockIdx.x % NHEAD;
    const int t = threadIdx.x;

    if (t < SEQ) {
        size_t row = ((size_t)(b * SEQ + t)) * DMODEL + (size_t)h * DKHEAD;
#pragma unroll
        for (int i = 0; i < 8; ++i) {           // 8 x uint4 = 64 bf16 (first half of row)
            ((uint4*)lq[t])[i] = ((const uint4*)(q + row))[i];
            ((uint4*)lk[t])[i] = ((const uint4*)(k + row))[i];
            ((uint4*)lv[t])[i] = ((const uint4*)(v + row))[i];
        }
        int tok = src[b * SEQ + t];
        mk[t] = (tok != 0 || t == 0) ? 0.0f : -1.0e9f;
    }
    __syncthreads();

    if (t < SEQ) {
        const float scale = 0.08838834764831845f;  // 1/sqrt(128)
        float sc[SEQ];
        float mx = -3.0e38f;
        for (int j = 0; j < SEQ; ++j) {
            float d = 0.f;
#pragma unroll
            for (int dd = 0; dd < DKHEAD; ++dd)
                d += (float)lq[t][dd] * (float)lk[j][dd];
            d = d * scale + mk[j];
            sc[j] = d;
            mx = fmaxf(mx, d);
        }
        float sum = 0.f;
        for (int j = 0; j < SEQ; ++j) {
            sc[j] = __expf(sc[j] - mx);
            sum += sc[j];
        }
        const float inv = 1.0f / sum;
        size_t orow = ((size_t)(b * SEQ + t)) * DMODEL + (size_t)h * DKHEAD;
        for (int dd = 0; dd < DKHEAD; ++dd) {
            float accv = 0.f;
            for (int j = 0; j < SEQ; ++j) accv += sc[j] * (float)lv[j][dd];
            o[orow + dd] = (bf16_t)(accv * inv);
        }
    }
}

// ---------------------------------------------------------------------------
// pooled = feats[:, 0, :]
// ---------------------------------------------------------------------------
__global__ void pooled_kernel(const float* __restrict__ feats,
                              float* __restrict__ pooled) {
    size_t i = (size_t)blockIdx.x * blockDim.x + threadIdx.x;
    if (i >= (size_t)BATCH * DMODEL) return;
    size_t b = i / DMODEL;
    size_t d = i % DMODEL;
    pooled[i] = feats[b * SEQ * DMODEL + d];
}

// ---------------------------------------------------------------------------
// Segment means + valid. One block per batch element; each thread owns 4
// strided columns exclusively. Zeroes its region of `means` first.
// ---------------------------------------------------------------------------
__global__ void means_kernel(const float* __restrict__ feats,
                             const int* __restrict__ src,
                             float* __restrict__ means,
                             float* __restrict__ validOut) {
    __shared__ int   seg[SEQ];
    __shared__ int   issep[SEQ];
    __shared__ float cnt[SEQ];
    __shared__ float validv[SEQ];

    const int b = blockIdx.x;
    const int t = threadIdx.x;

    if (t == 0) {
        int cum = 0;
        for (int j = 0; j < SEQ; ++j) {
            int tok = src[b * SEQ + j];
            int sp  = (tok == 1) ? 1 : 0;
            cum += sp;
            seg[j]   = cum - sp;
            issep[j] = sp;
        }
        for (int g2 = 0; g2 < SEQ; ++g2) cnt[g2] = 0.f;
        for (int j = 0; j < SEQ; ++j)
            if (!issep[j]) cnt[seg[j]] += 1.0f;
        int num_sep = cum;
        int last = src[b * SEQ + SEQ - 1];
        bool last_ok = (last != 0) && (last != 1);
        for (int g2 = 0; g2 < SEQ; ++g2) {
            bool vq = (g2 < num_sep) || ((g2 == num_sep) && last_ok);
            validv[g2] = vq ? 1.0f : 0.0f;
        }
    }
    __syncthreads();

    size_t base = (size_t)b * SEQ * DMODEL;
#pragma unroll
    for (int i = 0; i < 4; ++i) {
        int c = t + 256 * i;
        for (int g2 = 0; g2 < SEQ; ++g2)
            means[base + (size_t)g2 * DMODEL + c] = 0.0f;
        for (int j = 0; j < SEQ; ++j)
            if (!issep[j])
                means[base + (size_t)seg[j] * DMODEL + c] +=
                    feats[base + (size_t)j * DMODEL + c];
        for (int g2 = 0; g2 < SEQ; ++g2)
            means[base + (size_t)g2 * DMODEL + c] *=
                validv[g2] / fmaxf(cnt[g2], 1.0f);
    }
    if (t < SEQ) validOut[b * SEQ + t] = validv[t];
}

// ---------------------------------------------------------------------------
// Host launcher
// ---------------------------------------------------------------------------
extern "C" void kernel_launch(void* const* d_in, const int* in_sizes, int n_in,
                              void* d_out, int out_size, void* d_ws, size_t ws_size,
                              hipStream_t stream) {
    (void)in_sizes; (void)n_in; (void)out_size; (void)ws_size;

    const int*   src    = (const int*)  d_in[0];
    const float* embed  = (const float*)d_in[1];
    const float* Wq     = (const float*)d_in[2];
    const float* bq     = (const float*)d_in[3];
    const float* Wk     = (const float*)d_in[4];
    const float* bk     = (const float*)d_in[5];
    const float* Wv     = (const float*)d_in[6];
    const float* bv     = (const float*)d_in[7];
    const float* Wo     = (const float*)d_in[8];
    const float* bo     = (const float*)d_in[9];
    const float* ln1_g  = (const float*)d_in[10];
    const float* ln1_b  = (const float*)d_in[11];
    const float* W1     = (const float*)d_in[12];
    const float* b1     = (const float*)d_in[13];
    const float* W2     = (const float*)d_in[14];
    const float* b2     = (const float*)d_in[15];
    const float* ln2_g  = (const float*)d_in[16];
    const float* ln2_b  = (const float*)d_in[17];
    const float* lnf_g  = (const float*)d_in[18];
    const float* lnf_b  = (const float*)d_in[19];

    const size_t MD  = (size_t)MROWS * DMODEL;
    const size_t MF  = (size_t)MROWS * FFDIM;
    const size_t WDD = (size_t)NLAYER * DMODEL * DMODEL;
    const size_t WDF = (size_t)NLAYER * DMODEL * FFDIM;

    // workspace layout (transposed bf16 weights: WT[l][N][K])
    char* p = (char*)d_ws;
    float*  X    = (float*)p;        p += MD * sizeof(float);
    bf16_t* Y    = (bf16_t*)p;       p += MD * sizeof(bf16_t);
    bf16_t* SCR  = (bf16_t*)p;       p += MF * sizeof(bf16_t);  // qkv OR ffn-h
    bf16_t* WqT  = (bf16_t*)p;       p += WDD * sizeof(bf16_t);
    bf16_t* WkT  = (bf16_t*)p;       p += WDD * sizeof(bf16_t);
    bf16_t* WvT  = (bf16_t*)p;       p += WDD * sizeof(bf16_t);
    bf16_t* WoT  = (bf16_t*)p;       p += WDD * sizeof(bf16_t);
    bf16_t* W1T  = (bf16_t*)p;       p += WDF * sizeof(bf16_t);
    bf16_t* W2T  = (bf16_t*)p;       /* end */

    bf16_t* qb  = SCR;
    bf16_t* kb2 = SCR + MD;
    bf16_t* vb  = SCR + 2 * MD;

    float* feats  = (float*)d_out;
    float* pooled = feats + MD;
    float* meansO = pooled + (size_t)BATCH * DMODEL;
    float* validO = meansO + MD;

    // --- stage weights: transpose + convert to bf16 (per layer via grid.z)
    auto xcvt = [&](const float* W, bf16_t* WT, int K, int N) {
        hipLaunchKernelGGL(transpose_cvt_kernel, dim3(K / 32, N / 32, NLAYER),
                           dim3(32, 8), 0, stream, W, WT, K, N);
    };
    xcvt(Wq, WqT, DMODEL, DMODEL);
    xcvt(Wk, WkT, DMODEL, DMODEL);
    xcvt(Wv, WvT, DMODEL, DMODEL);
    xcvt(Wo, WoT, DMODEL, DMODEL);
    xcvt(W1, W1T, DMODEL, FFDIM);
    xcvt(W2, W2T, FFDIM,  DMODEL);

    // --- embedding + positional encoding
    hipLaunchKernelGGL(embed_kernel, dim3((unsigned)(MD / 256)), dim3(256), 0,
                       stream, src, embed, X);

    const dim3 gemmBlk(256);
    const dim3 gridD(DMODEL / 64, MROWS / 128);   // (16, 120)
    const dim3 gridF(FFDIM  / 64, MROWS / 128);   // (64, 120)

    for (int l = 0; l < NLAYER; ++l) {
        const size_t wDD = (size_t)l * DMODEL * DMODEL;
        const size_t wDF = (size_t)l * DMODEL * FFDIM;

        hipLaunchKernelGGL((ln_kernel<false>), dim3(MROWS), dim3(256), 0, stream,
                           X, ln1_g + l * DMODEL, ln1_b + l * DMODEL, Y, nullptr);

        hipLaunchKernelGGL((gemm_wmma_bf16_kernel<4>), gridD, gemmBlk, 0, stream,
                           Y, WqT + wDD, bq + l * DMODEL, nullptr, nullptr, qb,
                           MROWS, DMODEL, DMODEL);
        hipLaunchKernelGGL((gemm_wmma_bf16_kernel<4>), gridD, gemmBlk, 0, stream,
                           Y, WkT + wDD, bk + l * DMODEL, nullptr, nullptr, kb2,
                           MROWS, DMODEL, DMODEL);
        hipLaunchKernelGGL((gemm_wmma_bf16_kernel<4>), gridD, gemmBlk, 0, stream,
                           Y, WvT + wDD, bv + l * DMODEL, nullptr, nullptr, vb,
                           MROWS, DMODEL, DMODEL);

        hipLaunchKernelGGL(attn_kernel, dim3(BATCH * NHEAD), dim3(64), 0, stream,
                           qb, kb2, vb, src, Y);

        hipLaunchKernelGGL((gemm_wmma_bf16_kernel<2>), gridD, gemmBlk, 0, stream,
                           Y, WoT + wDD, bo + l * DMODEL, X, X, nullptr,
                           MROWS, DMODEL, DMODEL);

        hipLaunchKernelGGL((ln_kernel<false>), dim3(MROWS), dim3(256), 0, stream,
                           X, ln2_g + l * DMODEL, ln2_b + l * DMODEL, Y, nullptr);

        hipLaunchKernelGGL((gemm_wmma_bf16_kernel<4 | 1>), gridF, gemmBlk, 0,
                           stream, Y, W1T + wDF, b1 + l * FFDIM,
                           nullptr, nullptr, SCR, MROWS, FFDIM, DMODEL);

        hipLaunchKernelGGL((gemm_wmma_bf16_kernel<2>), gridD, gemmBlk, 0, stream,
                           SCR, W2T + wDF, b2 + l * DMODEL, X, X, nullptr,
                           MROWS, DMODEL, FFDIM);
    }

    hipLaunchKernelGGL((ln_kernel<true>), dim3(MROWS), dim3(256), 0, stream,
                       X, lnf_g, lnf_b, nullptr, feats);

    hipLaunchKernelGGL(pooled_kernel,
                       dim3((unsigned)(((size_t)BATCH * DMODEL) / 256)),
                       dim3(256), 0, stream, feats, pooled);

    hipLaunchKernelGGL(means_kernel, dim3(BATCH), dim3(256), 0, stream,
                       feats, src, meansO, validO);
}
